// DSCNMP_10282151707326
// MI455X (gfx1250) — matrix operand order repaired and checked by reading, not compiled, not run on hardware
//
#include <hip/hip_runtime.h>

#define NN 100000
#define EE 600000
#define GG 1000
#define HH 128

typedef _Float16 v8h  __attribute__((ext_vector_type(8)));
typedef _Float16 v16h __attribute__((ext_vector_type(16)));
typedef float    v8f  __attribute__((ext_vector_type(8)));

// ---------------------------------------------------------------------------
// WMMA GEMM: Y[nrows,128] = epi( X[nrows,128] @ W[128,128] + bias )
// epi = ReLU, optionally followed by eval-mode BatchNorm.
// Block: 128 threads = 4 wave32; each wave computes a 16-row strip x 128 cols
// as 8 tiles of 16x16 via v_wmma_f32_16x16x32_f16 (4 k-steps).
// ---------------------------------------------------------------------------
__global__ __launch_bounds__(128) void gemm128_wmma(
    const float* __restrict__ X, const float* __restrict__ W,
    const float* __restrict__ bias,
    const float* __restrict__ bng, const float* __restrict__ bnb,
    const float* __restrict__ bnrm, const float* __restrict__ bnrv,
    float* __restrict__ Y, int nrows, int do_bn)
{
    __shared__ _Float16 sW[HH * HH];   // transposed: sW[n*128+k] = W[k][n]  (32 KB)
    __shared__ _Float16 sX[64 * HH];   // row tile                           (16 KB)

    const int tid  = threadIdx.x;
    const int lane = tid & 31;
    const int wave = tid >> 5;               // m-tile within 64-row block
    const int rowBase = blockIdx.x * 64;

    // Stage weights (f32 -> f16, transposed into LDS)
    for (int idx = tid; idx < HH * HH; idx += 128) {
        int k = idx >> 7, n = idx & 127;
        sW[n * HH + k] = (_Float16)W[idx];
    }
    // Stage activation rows (OOB rows -> 0)
    for (int idx = tid; idx < 64 * HH; idx += 128) {
        int r = idx >> 7, k = idx & 127;
        int row = rowBase + r;
        float v = (row < nrows) ? X[row * HH + k] : 0.0f;
        sX[idx] = (_Float16)v;
    }
    __syncthreads();

    const int m  = lane & 15;   // row within tile (A), col within tile (B/C)
    const int hi = lane >> 4;   // half-wave selector per ISA frag layouts

    v8f c[8] = {};              // 8 N-tiles of 16x16 f32 accumulators

    const _Float16* xrow = sX + (wave * 16 + m) * HH;

#pragma unroll
    for (int ks = 0; ks < 4; ++ks) {
        const int k0 = ks * 32;
        // A fragment (16x32 f16): lanes 0-15 K=[k0..k0+15];
        // lanes 16-31 K=[k0+16..k0+31] — two aligned 16B LDS loads.
        v8h alo = *(const v8h*)(xrow + k0 + 16 * hi);
        v8h ahi = *(const v8h*)(xrow + k0 + 16 * hi + 8);
        v16h a = __builtin_shufflevector(alo, ahi,
                    0,1,2,3,4,5,6,7,8,9,10,11,12,13,14,15);
#pragma unroll
        for (int nt = 0; nt < 8; ++nt) {
            // B fragment (32x16 f16): lane holds col n, 16 contiguous K values.
            const _Float16* wrow = sW + (nt * 16 + m) * HH + k0 + 16 * hi;
            v8h blo = *(const v8h*)(wrow);
            v8h bhi = *(const v8h*)(wrow + 8);
            v16h b = __builtin_shufflevector(blo, bhi,
                        0,1,2,3,4,5,6,7,8,9,10,11,12,13,14,15);
            c[nt] = __builtin_amdgcn_wmma_f32_16x16x32_f16(
                        false, a, false, b, (short)0, c[nt], false, false);
        }
    }

    // Epilogue: bias + ReLU (+ BN), C layout: VGPR r -> row r + 8*hi
#pragma unroll
    for (int nt = 0; nt < 8; ++nt) {
        const int col = nt * 16 + m;
        const float bs = bias[col];
        float scale = 1.0f, shift = 0.0f;
        if (do_bn) {
            float s = bng[col] * rsqrtf(bnrv[col] + 1e-5f);
            scale = s;
            shift = bnb[col] - bnrm[col] * s;
        }
#pragma unroll
        for (int r = 0; r < 8; ++r) {
            int row = rowBase + wave * 16 + hi * 8 + r;
            if (row < nrows) {
                float v = c[nt][r] + bs;
                v = v > 0.0f ? v : 0.0f;
                Y[row * HH + col] = v * scale + shift;
            }
        }
    }
}

// ---------------------------------------------------------------------------
// Helper kernels
// ---------------------------------------------------------------------------
__global__ void k_zero(float* p, int n) {
    int t = blockIdx.x * blockDim.x + threadIdx.x;
    if (t < n) p[t] = 0.0f;
}

__global__ void k_copy4(const float* __restrict__ s, float* __restrict__ d, int n4) {
    int t = blockIdx.x * blockDim.x + threadIdx.x;
    if (t < n4) ((float4*)d)[t] = ((const float4*)s)[t];
}

// agg[dst] += pos[src] (C=2)
__global__ void k_scatter_pos(const int* __restrict__ ei, const float* __restrict__ pos,
                              float* __restrict__ agg, int ne) {
    int e = blockIdx.x * blockDim.x + threadIdx.x;
    if (e >= ne) return;
    int s = ei[e], d = ei[ne + e];
    unsafeAtomicAdd(&agg[d * 2 + 0], pos[s * 2 + 0]);
    unsafeAtomicAdd(&agg[d * 2 + 1], pos[s * 2 + 1]);
}

// agg[dst,:] += x[src,:] (H=128), thread = (edge, 4-feature chunk)
__global__ void k_scatter_h(const int* __restrict__ ei, const float* __restrict__ x,
                            float* __restrict__ agg, int ne) {
    int t = blockIdx.x * blockDim.x + threadIdx.x;
    if (t >= ne * 32) return;
    int e = t >> 5, c = (t & 31) * 4;
    int s = ei[e], d = ei[ne + e];
    const float* xs = x + (size_t)s * HH + c;
    float* ad = agg + (size_t)d * HH + c;
    unsafeAtomicAdd(ad + 0, xs[0]);
    unsafeAtomicAdd(ad + 1, xs[1]);
    unsafeAtomicAdd(ad + 2, xs[2]);
    unsafeAtomicAdd(ad + 3, xs[3]);
}

__global__ void k_pool2(const float* __restrict__ pos, const int* __restrict__ batch,
                        float* __restrict__ pool, int n) {
    int i = blockIdx.x * blockDim.x + threadIdx.x;
    if (i >= n) return;
    int g = batch[i];
    unsafeAtomicAdd(&pool[g * 2 + 0], pos[i * 2 + 0]);
    unsafeAtomicAdd(&pool[g * 2 + 1], pos[i * 2 + 1]);
}

__global__ void k_pool128(const float* __restrict__ x, const int* __restrict__ batch,
                          float* __restrict__ pool, int n) {
    int t = blockIdx.x * blockDim.x + threadIdx.x;
    if (t >= n * 32) return;
    int i = t >> 5, c = (t & 31) * 4;
    int g = batch[i];
    const float* xs = x + (size_t)i * HH + c;
    float* pd = pool + (size_t)g * HH + c;
    unsafeAtomicAdd(pd + 0, xs[0]);
    unsafeAtomicAdd(pd + 1, xs[1]);
    unsafeAtomicAdd(pd + 2, xs[2]);
    unsafeAtomicAdd(pd + 3, xs[3]);
}

// Y[i,j] = relu(X2[i,0]*W[0,j] + X2[i,1]*W[1,j] + b[j])   (K=2 linear)
__global__ void k_lin2_relu(const float* __restrict__ X2, const float* __restrict__ W,
                            const float* __restrict__ b, float* __restrict__ Y, int n) {
    int t = blockIdx.x * blockDim.x + threadIdx.x;
    if (t >= n * HH) return;
    int i = t >> 7, j = t & 127;
    float v = X2[i * 2] * W[j] + X2[i * 2 + 1] * W[HH + j] + b[j];
    Y[t] = v > 0.0f ? v : 0.0f;
}

// fc1: BN(relu(P@Wf1 + bf1)), K=2
__global__ void k_fc1(const float* __restrict__ P, const float* __restrict__ W,
                      const float* __restrict__ b, const float* __restrict__ g,
                      const float* __restrict__ be, const float* __restrict__ rm,
                      const float* __restrict__ rv, float* __restrict__ Y, int n) {
    int t = blockIdx.x * blockDim.x + threadIdx.x;
    if (t >= n * HH) return;
    int i = t >> 7, j = t & 127;
    float v = P[i * 2] * W[j] + P[i * 2 + 1] * W[HH + j] + b[j];
    v = v > 0.0f ? v : 0.0f;
    float s = g[j] * rsqrtf(rv[j] + 1e-5f);
    Y[t] = (v - rm[j]) * s + be[j];
}

__global__ void k_add2(const float* a, const float* b, float* y, int n) {
    int t = blockIdx.x * blockDim.x + threadIdx.x;
    if (t < n) y[t] = a[t] + b[t];
}

__global__ void k_add3(const float* a, const float* b, const float* c, float* y, int n) {
    int t = blockIdx.x * blockDim.x + threadIdx.x;
    if (t < n) y[t] = a[t] + b[t] + c[t];
}

// classifier: h = prelu(BN(x@Wc1 + bc1)); out = h@Wc2 + bc2   (one block / graph)
__global__ __launch_bounds__(64) void k_cls(
    const float* __restrict__ Xg, const float* __restrict__ Wc1,
    const float* __restrict__ bc1, const float* __restrict__ gc,
    const float* __restrict__ bec, const float* __restrict__ rmc,
    const float* __restrict__ rvc, const float* __restrict__ a_prelu,
    const float* __restrict__ Wc2, const float* __restrict__ bc2,
    float* __restrict__ out) {
    __shared__ float xr[128];
    __shared__ float h[64];
    int gi = blockIdx.x, j = threadIdx.x;
    xr[j]      = Xg[gi * HH + j];
    xr[j + 64] = Xg[gi * HH + 64 + j];
    __syncthreads();
    float acc = bc1[j];
#pragma unroll 8
    for (int k = 0; k < 128; ++k) acc += xr[k] * Wc1[k * 64 + j];
    float s = gc[j] * rsqrtf(rvc[j] + 1e-5f);
    float v = (acc - rmc[j]) * s + bec[j];
    float ap = a_prelu[0];
    h[j] = v > 0.0f ? v : ap * v;
    __syncthreads();
    if (j < 10) {
        float o = bc2[j];
#pragma unroll 8
        for (int k = 0; k < 64; ++k) o += h[k] * Wc2[k * 10 + j];
        out[gi * 10 + j] = o;
    }
}

// ---------------------------------------------------------------------------
extern "C" void kernel_launch(void* const* d_in, const int* in_sizes, int n_in,
                              void* d_out, int out_size, void* d_ws, size_t ws_size,
                              hipStream_t stream) {
    const float* pos   = (const float*)d_in[0];
    const int*   ei    = (const int*)d_in[1];
    const int*   batch = (const int*)d_in[2];
    const float* W1a = (const float*)d_in[3];  const float* b1a = (const float*)d_in[4];
    const float* W1b = (const float*)d_in[5];  const float* b1b = (const float*)d_in[6];
    const float* n1g = (const float*)d_in[7];  const float* n1b = (const float*)d_in[8];
    const float* n1rm = (const float*)d_in[9]; const float* n1rv = (const float*)d_in[10];
    const float* W2a = (const float*)d_in[11]; const float* b2a = (const float*)d_in[12];
    const float* W2b = (const float*)d_in[13]; const float* b2b = (const float*)d_in[14];
    const float* n2g = (const float*)d_in[15]; const float* n2b = (const float*)d_in[16];
    const float* n2rm = (const float*)d_in[17]; const float* n2rv = (const float*)d_in[18];
    const float* Wf1 = (const float*)d_in[19]; const float* bf1 = (const float*)d_in[20];
    const float* f1g = (const float*)d_in[21]; const float* f1b = (const float*)d_in[22];
    const float* f1rm = (const float*)d_in[23]; const float* f1rv = (const float*)d_in[24];
    const float* Wf2 = (const float*)d_in[25]; const float* bf2 = (const float*)d_in[26];
    const float* f2g = (const float*)d_in[27]; const float* f2b = (const float*)d_in[28];
    const float* f2rm = (const float*)d_in[29]; const float* f2rv = (const float*)d_in[30];
    const float* Wc1 = (const float*)d_in[31]; const float* bc1 = (const float*)d_in[32];
    const float* gc  = (const float*)d_in[33]; const float* bec = (const float*)d_in[34];
    const float* rmc = (const float*)d_in[35]; const float* rvc = (const float*)d_in[36];
    const float* a_prelu = (const float*)d_in[37];
    const float* Wc2 = (const float*)d_in[38]; const float* bc2 = (const float*)d_in[39];
    float* out = (float*)d_out;

    // workspace carve-up (floats)
    float* bufA    = (float*)d_ws;            // h1 / h2        [N,128]
    float* bufB    = bufA + (size_t)NN * HH;  // x1             [N,128]
    float* bufC    = bufB + (size_t)NN * HH;  // agg2 / x2      [N,128]
    float* aggpos  = bufC + (size_t)NN * HH;  // [N,2]
    float* poolpos = aggpos + (size_t)NN * 2; // [G,2]
    float* poolx1  = poolpos + GG * 2;        // [G,128]
    float* poolx2  = poolx1 + GG * HH;        // [G,128]
    float* x0g     = poolx2 + GG * HH;
    float* x1g     = x0g + GG * HH;
    float* fcin    = x1g + GG * HH;
    float* x2g     = fcin + GG * HH;

    const int TB = 256;
    const int gemmBlocksN = (NN + 63) / 64;
    const int gemmBlocksG = (GG + 63) / 64;

    // 0) zero pool accumulators (poolpos..poolx2 contiguous)
    int nzero = GG * (2 + 2 * HH);
    k_zero<<<(nzero + TB - 1) / TB, TB, 0, stream>>>(poolpos, nzero);

    // 1) gin1 aggregation: aggpos = pos + scatter(pos)
    k_copy4<<<((NN * 2 / 4) + TB - 1) / TB, TB, 0, stream>>>(pos, aggpos, NN * 2 / 4);
    k_scatter_pos<<<(EE + TB - 1) / TB, TB, 0, stream>>>(ei, pos, aggpos, EE);

    // 2) gin1 MLP: h1 = relu(aggpos@W1a+b1a);  x1 = BN(relu(h1@W1b+b1b))
    k_lin2_relu<<<((NN * HH) + TB - 1) / TB, TB, 0, stream>>>(aggpos, W1a, b1a, bufA, NN);
    gemm128_wmma<<<gemmBlocksN, 128, 0, stream>>>(bufA, W1b, b1b,
        n1g, n1b, n1rm, n1rv, bufB, NN, 1);

    // 3) pools of pos and x1
    k_pool2<<<(NN + TB - 1) / TB, TB, 0, stream>>>(pos, batch, poolpos, NN);
    k_pool128<<<((NN * 32) + TB - 1) / TB, TB, 0, stream>>>(bufB, batch, poolx1, NN);

    // 4) gin2 aggregation: agg2 = x1 + scatter(x1)
    k_copy4<<<((NN * HH / 4) + TB - 1) / TB, TB, 0, stream>>>(bufB, bufC, NN * HH / 4);
    k_scatter_h<<<((EE * 32) + TB - 1) / TB, TB, 0, stream>>>(ei, bufB, bufC, EE);

    // 5) gin2 MLP: h2 = relu(agg2@W2a+b2a);  x2 = BN(relu(h2@W2b+b2b))
    gemm128_wmma<<<gemmBlocksN, 128, 0, stream>>>(bufC, W2a, b2a,
        b2a, b2a, b2a, b2a, bufA, NN, 0);
    gemm128_wmma<<<gemmBlocksN, 128, 0, stream>>>(bufA, W2b, b2b,
        n2g, n2b, n2rm, n2rv, bufC, NN, 1);

    // 6) pool of x2
    k_pool128<<<((NN * 32) + TB - 1) / TB, TB, 0, stream>>>(bufC, batch, poolx2, NN);

    // 7) graph-level fc chain
    k_fc1<<<((GG * HH) + TB - 1) / TB, TB, 0, stream>>>(poolpos, Wf1, bf1,
        f1g, f1b, f1rm, f1rv, x0g, GG);
    k_add2<<<((GG * HH) + TB - 1) / TB, TB, 0, stream>>>(x0g, poolx1, fcin, GG * HH);
    gemm128_wmma<<<gemmBlocksG, 128, 0, stream>>>(fcin, Wf2, bf2,
        f2g, f2b, f2rm, f2rv, x1g, GG, 1);
    k_add3<<<((GG * HH) + TB - 1) / TB, TB, 0, stream>>>(x0g, x1g, poolx2, fcin, GG * HH);
    gemm128_wmma<<<gemmBlocksG, 128, 0, stream>>>(fcin, Wf2, bf2,
        f2g, f2b, f2rm, f2rv, x2g, GG, 1);

    // 8) classifier
    k_cls<<<GG, 64, 0, stream>>>(x2g, Wc1, bc1, gc, bec, rmc, rvc, a_prelu,
                                 Wc2, bc2, out);
}
